// LocalRCT_6751688589627
// MI455X (gfx1250) — compile-verified, matrix operand order checked by reading, not run on hardware
//
#include <hip/hip_runtime.h>
#include <hip/hip_bf16.h>
#include <math.h>

// ---------------------------------------------------------------------------
// CDNA5 (gfx1250) wave32 WMMA implementation of LocalRCT.
// conv blocks: implicit GEMM on padded [C][34][34] inputs, offset-table im2col,
// double-buffered LDS B tiles, 4 WMMAs per wave per K-step.
// attention: per-cell WMMA scores/softmax/output with conflict-free LDS.
// ---------------------------------------------------------------------------

typedef __attribute__((ext_vector_type(16))) _Float16 v16h;
typedef __attribute__((ext_vector_type(8)))  _Float16 v8h;
typedef __attribute__((ext_vector_type(8)))  float    v8f;

#define WMMA16x16x32(a, b, c) \
  __builtin_amdgcn_wmma_f32_16x16x32_f16(false, (a), false, (b), (short)0, (c), false, false)

// 16x32 f16 fragment load (A row-major, or B from a column-major [N][K] stage).
// ISA layout: lane&15 = row; lane>>4 selects K-halves {0..7,16..23} / {8..15,24..31}.
__device__ __forceinline__ v16h frag_ld(const _Float16* base, int stride) {
  int lane = threadIdx.x & 31;
  int r    = lane & 15;
  int kb   = (lane >> 4) << 3;
  const _Float16* p = base + r * stride + kb;
  v8h lo = *(const v8h*)(p);
  v8h hi = *(const v8h*)(p + 16);
  v16h f;
#pragma unroll
  for (int i = 0; i < 8; ++i) { f[i] = lo[i]; f[i + 8] = hi[i]; }
  return f;
}
// C/D fragment: element i of lane L -> row (i + 8*(L>>4)), col (L&15).

// ---------------------------------------------------------------------------
// small prep kernels
// ---------------------------------------------------------------------------
__global__ void cvt_f32_f16(const float* __restrict__ src,
                            _Float16* __restrict__ dst, int n) {
  int i = blockIdx.x * blockDim.x + threadIdx.x;
  if (i < n) dst[i] = (_Float16)src[i];
}

// fp32 [C][32][32] -> f16 zero-padded [C][34][34]
__global__ void cvt_pad_f16(const float* __restrict__ src,
                            _Float16* __restrict__ dst, int n) {
  int e = blockIdx.x * blockDim.x + threadIdx.x;
  if (e >= n) return;
  int c = e / 1156, p = e - c * 1156;
  int yy = p / 34, xx = p - yy * 34;
  _Float16 v = (_Float16)0.0f;
  if (yy >= 1 && yy <= 32 && xx >= 1 && xx <= 32)
    v = (_Float16)src[(c * 32 + (yy - 1)) * 32 + (xx - 1)];
  dst[e] = v;
}

__global__ void fill_zero_f16(_Float16* __restrict__ p, int n) {
  int i = blockIdx.x * blockDim.x + threadIdx.x;
  if (i < n) p[i] = (_Float16)0.0f;
}

// k (0..1151) -> padded-input offset ci*1156 + kh*34 + kw
__global__ void make_offt(int* __restrict__ t) {
  int k = blockIdx.x * blockDim.x + threadIdx.x;
  if (k < 1152) {
    int ci = k / 9, r = k - ci * 9, kh = r / 3, kw = r - kh * 3;
    t[k] = ci * 1156 + kh * 34 + kw;
  }
}

// ---------------------------------------------------------------------------
// conv3x3 (128 -> 128) + BN + SiLU. M=128, K=1152, N=8192.
// Block: 8 waves, tile 128x64; wave w owns co-tile w, 4 n-tiles.
// in: padded [8*128][34][34] f16, out: padded [8*128][34][34] f16 (interior).
// ---------------------------------------------------------------------------
__global__ void conv1_bn_silu(const _Float16* __restrict__ in_pad,
                              const _Float16* __restrict__ w16,   // [128][1152]
                              const int* __restrict__ g_offt,
                              const float* __restrict__ bias,
                              const float* __restrict__ gamma,
                              const float* __restrict__ beta,
                              const float* __restrict__ mean,
                              const float* __restrict__ var,
                              _Float16* __restrict__ out_pad) {
  __shared__ __align__(16) _Float16 Bt[2][64][32];
  __shared__ int offt[1152];
  const int tid  = threadIdx.x;
  const int wv   = tid >> 5;
  const int lane = tid & 31;
  const int col  = lane & 15;
  const int rhalf = (lane >> 4) << 3;
  const int n0 = blockIdx.x * 64;

  // per-thread fill geometry (fixed across K loop)
  const int nl  = tid >> 2;                 // 0..63
  const int kks = (tid & 3) << 3;           // 0,8,16,24
  {
    for (int k = tid; k < 1152; k += 256) offt[k] = g_offt[k];
  }
  const int nf  = n0 + nl;
  const int bf  = nf >> 10, remf = nf & 1023, yf = remf >> 5, xf = remf & 31;
  const _Float16* inbase = in_pad + (size_t)bf * 128 * 1156 + yf * 34 + xf;
  __syncthreads();

  // prologue fill buffer 0
  {
    v8h vv;
#pragma unroll
    for (int s = 0; s < 8; ++s) vv[s] = inbase[offt[kks + s]];
    *(v8h*)&Bt[0][nl][kks] = vv;
  }
  __syncthreads();

  v8f acc[4] = {};
  for (int it = 0; it < 36; ++it) {
    const int cur = it & 1;
    if (it < 35) {
      const int k0n = (it + 1) * 32;
      __builtin_prefetch(w16 + (wv * 16 + col) * 1152 + k0n, 0, 0);
      v8h vv;
#pragma unroll
      for (int s = 0; s < 8; ++s) vv[s] = inbase[offt[k0n + kks + s]];
      *(v8h*)&Bt[1 - cur][nl][kks] = vv;
    }
    v16h a = frag_ld(w16 + (wv * 16) * 1152 + it * 32, 1152);
#pragma unroll
    for (int nt = 0; nt < 4; ++nt) {
      v16h bfr = frag_ld(&Bt[cur][nt * 16][0], 32);
      acc[nt] = WMMA16x16x32(a, bfr, acc[nt]);
    }
    __syncthreads();
  }

  // epilogue: bias + BN + SiLU -> padded f16 store
  float inv8[8], sh8[8], b8[8];
  int co8[8];
#pragma unroll
  for (int i = 0; i < 8; ++i) {
    int co = wv * 16 + i + rhalf;
    co8[i] = co;
    float inv = gamma[co] * rsqrtf(var[co] + 1e-5f);
    inv8[i] = inv;
    sh8[i]  = beta[co] - mean[co] * inv;
    b8[i]   = bias[co];
  }
#pragma unroll
  for (int nt = 0; nt < 4; ++nt) {
    int n = n0 + nt * 16 + col;
    int b = n >> 10, rem = n & 1023, y = rem >> 5, x = rem & 31;
#pragma unroll
    for (int i = 0; i < 8; ++i) {
      float h = (acc[nt][i] + b8[i]) * inv8[i] + sh8[i];
      float s = h / (1.0f + __expf(-h));      // SiLU
      out_pad[((size_t)(b * 128 + co8[i])) * 1156 + (y + 1) * 34 + (x + 1)] = (_Float16)s;
    }
  }
}

// ---------------------------------------------------------------------------
// conv3x3 (128 -> Co) + bias. Co = 256 (r) or 48 (t). Output unpadded.
// ---------------------------------------------------------------------------
__global__ void conv2_bias(const _Float16* __restrict__ in_pad,
                           const _Float16* __restrict__ w16,   // [Co][1152]
                           const int* __restrict__ g_offt,
                           const float* __restrict__ bias,
                           _Float16* __restrict__ out16,       // [8][Co][32][32]
                           int Co) {
  __shared__ __align__(16) _Float16 Bt[2][64][32];
  __shared__ int offt[1152];
  const int tid  = threadIdx.x;
  const int wv   = tid >> 5;
  const int lane = tid & 31;
  const int col  = lane & 15;
  const int rhalf = (lane >> 4) << 3;
  const int mtile = blockIdx.y * 8 + wv;
  const bool active = (mtile * 16) < Co;
  const int n0 = blockIdx.x * 64;

  const int nl  = tid >> 2;
  const int kks = (tid & 3) << 3;
  for (int k = tid; k < 1152; k += 256) offt[k] = g_offt[k];
  const int nf  = n0 + nl;
  const int bf  = nf >> 10, remf = nf & 1023, yf = remf >> 5, xf = remf & 31;
  const _Float16* inbase = in_pad + (size_t)bf * 128 * 1156 + yf * 34 + xf;
  __syncthreads();

  {
    v8h vv;
#pragma unroll
    for (int s = 0; s < 8; ++s) vv[s] = inbase[offt[kks + s]];
    *(v8h*)&Bt[0][nl][kks] = vv;
  }
  __syncthreads();

  v8f acc[4] = {};
  for (int it = 0; it < 36; ++it) {
    const int cur = it & 1;
    if (it < 35) {
      const int k0n = (it + 1) * 32;
      v8h vv;
#pragma unroll
      for (int s = 0; s < 8; ++s) vv[s] = inbase[offt[k0n + kks + s]];
      *(v8h*)&Bt[1 - cur][nl][kks] = vv;
    }
    if (active) {
      v16h a = frag_ld(w16 + (mtile * 16) * 1152 + it * 32, 1152);
#pragma unroll
      for (int nt = 0; nt < 4; ++nt) {
        v16h bfr = frag_ld(&Bt[cur][nt * 16][0], 32);
        acc[nt] = WMMA16x16x32(a, bfr, acc[nt]);
      }
    }
    __syncthreads();
  }

  if (active) {
#pragma unroll
    for (int nt = 0; nt < 4; ++nt) {
      int n = n0 + nt * 16 + col;
      int b = n >> 10, rem = n & 1023, y = rem >> 5, x = rem & 31;
#pragma unroll
      for (int i = 0; i < 8; ++i) {
        int co = mtile * 16 + i + rhalf;
        out16[((b * Co + co) * 32 + y) * 32 + x] = (_Float16)(acc[i == i ? i : i] + 0.0f, acc[nt][i] + bias[co]);
      }
    }
  }
}

// ---------------------------------------------------------------------------
// Per-grid-cell corner attention. One block (8 waves) per (b, i, j) cell.
//  scores[256x64] = fp[256x16(->32)] . r4[16(->32)x64] * 0.25
//  att = softmax(scores); y[256x3] = att[256x64] . t4^T[64x3(->16)]
// LDS strides padded for bank-conflict-free softmax (66 f32 / 72 f16).
// ---------------------------------------------------------------------------
#define SC_STR  66
#define ATT_STR 72
__global__ void attention_cell(const float* __restrict__ f,      // [8][16][496][496]
                               const _Float16* __restrict__ r16, // [8][256][32][32]
                               const _Float16* __restrict__ t16, // [8][48][32][32]
                               float* __restrict__ out) {        // [8][3][496][496]
  extern __shared__ char smem[];
  float*    sc   = (float*)(smem);                               // 256*66*4 = 67584
  _Float16* fpA  = (_Float16*)(smem + 67584);                    // 256*32*2 = 16384
  _Float16* attA = (_Float16*)(smem + 67584 + 16384);            // 256*72*2 = 36864
  _Float16* r4t  = (_Float16*)(smem + 67584 + 16384 + 36864);    // 64*32*2  =  4096
  _Float16* t4t  = (_Float16*)(smem + 67584 + 16384 + 36864 + 4096); // 16*64*2 = 2048

  const int tid  = threadIdx.x;
  const int wv   = tid >> 5;
  const int lane = tid & 31;
  const int col  = lane & 15;
  const int rhalf = (lane >> 4) << 3;

  int blk = blockIdx.x;
  int b   = blk / 961;
  int rem = blk - b * 961;
  int gi  = rem / 31, gj = rem - gi * 31;

  // warm the r/t gather footprint while the fp stage runs
  __builtin_prefetch(r16 + (((size_t)b * 256) * 32 + gi) * 32 + gj, 0, 0);
  __builtin_prefetch(t16 + (((size_t)b * 48) * 32 + gi) * 32 + gj, 0, 0);

  // ---- stage fp patch (256 pos x 16 ch, K padded to 32) ----
  {
    int p = tid, py = p >> 4, px = p & 15;
    const float* fb = f + ((size_t)b * 16) * 496 * 496 +
                      (size_t)(gi * 16 + py) * 496 + (gj * 16 + px);
#pragma unroll
    for (int c = 0; c < 16; ++c) {
      fpA[p * 32 + c]      = (_Float16)fb[(size_t)c * 496 * 496];
      fpA[p * 32 + 16 + c] = (_Float16)0.0f;
    }
  }
  // ---- stage r4^T: [n=q*16+l][k=c], K padded to 32 ----
#pragma unroll
  for (int s = 0; s < 8; ++s) {
    int e = tid * 8 + s;
    int n = e >> 5, k = e & 31;
    int q = n >> 4, l = n & 15;
    int yy = gi + (q & 1), xx = gj + ((q >> 1) & 1);
    _Float16 v = (_Float16)0.0f;
    if (k < 16) v = r16[((b * 256 + k * 16 + l) * 32 + yy) * 32 + xx];
    r4t[n * 32 + k] = v;
  }
  // ---- stage t4^T: [d (pad to 16)][k=q*16+l] ----
#pragma unroll
  for (int s = 0; s < 4; ++s) {
    int e = tid * 4 + s;
    int d = e >> 6, k = e & 63;
    int q = k >> 4, l = k & 15;
    int yy = gi + (q & 1), xx = gj + ((q >> 1) & 1);
    _Float16 v = (_Float16)0.0f;
    if (d < 3) v = t16[((b * 48 + d * 16 + l) * 32 + yy) * 32 + xx];
    t4t[d * 64 + k] = v;
  }
  __syncthreads();

  // ---- scores: 16 m-tiles x 4 n-tiles, 8 WMMA per wave ----
#pragma unroll
  for (int s = 0; s < 8; ++s) {
    int tt = wv * 8 + s;
    int mt = tt >> 2, nt = tt & 3;
    v16h a   = frag_ld(fpA + mt * 16 * 32, 32);
    v16h bfr = frag_ld(r4t + nt * 16 * 32, 32);
    v8f acc = {};
    acc = WMMA16x16x32(a, bfr, acc);
#pragma unroll
    for (int i = 0; i < 8; ++i) {
      int m = mt * 16 + i + rhalf;
      int n = nt * 16 + col;
      sc[m * SC_STR + n] = acc[i] * 0.25f;   // 1/sqrt(C), C=16
    }
  }
  __syncthreads();

  // ---- softmax over n (64), one row per thread; f16 att ----
  {
    float mx = -1e30f;
#pragma unroll
    for (int k = 0; k < 64; ++k) mx = fmaxf(mx, sc[tid * SC_STR + k]);
    float sum = 0.0f;
#pragma unroll
    for (int k = 0; k < 64; ++k) {
      float e = __expf(sc[tid * SC_STR + k] - mx);
      sc[tid * SC_STR + k] = e;
      sum += e;
    }
    float rinv = 1.0f / sum;
#pragma unroll
    for (int k = 0; k < 64; ++k)
      attA[tid * ATT_STR + k] = (_Float16)(sc[tid * SC_STR + k] * rinv);
  }
  __syncthreads();

  // ---- y = att . t4^T : 16 m-tiles, 2 per wave, K=64 (2 steps) ----
#pragma unroll
  for (int s = 0; s < 2; ++s) {
    int mt = wv * 2 + s;
    v8f acc = {};
#pragma unroll
    for (int k0 = 0; k0 < 64; k0 += 32) {
      v16h a   = frag_ld(attA + mt * 16 * ATT_STR + k0, ATT_STR);
      v16h bfr = frag_ld(t4t + k0, 64);
      acc = WMMA16x16x32(a, bfr, acc);
    }
    if (col < 3) {
#pragma unroll
      for (int i = 0; i < 8; ++i) {
        int p = mt * 16 + i + rhalf;
        int py = p >> 4, px = p & 15;
        out[((size_t)(b * 3 + col) * 496 + gi * 16 + py) * 496 + gj * 16 + px] = acc[i];
      }
    }
  }
}

// ---------------------------------------------------------------------------
// Host-side launch
// ---------------------------------------------------------------------------
extern "C" void kernel_launch(void* const* d_in, const int* in_sizes, int n_in,
                              void* d_out, int out_size, void* d_ws, size_t ws_size,
                              hipStream_t stream) {
  const float* f        = (const float*)d_in[0];
  const float* features = (const float*)d_in[1];
  const float* wR1 = (const float*)d_in[2];
  const float* bR1 = (const float*)d_in[3];
  const float* gR  = (const float*)d_in[4];
  const float* beR = (const float*)d_in[5];
  const float* mR  = (const float*)d_in[6];
  const float* vR  = (const float*)d_in[7];
  const float* wR2 = (const float*)d_in[8];
  const float* bR2 = (const float*)d_in[9];
  const float* wT1 = (const float*)d_in[10];
  const float* bT1 = (const float*)d_in[11];
  const float* gT  = (const float*)d_in[12];
  const float* beT = (const float*)d_in[13];
  const float* mT  = (const float*)d_in[14];
  const float* vT  = (const float*)d_in[15];
  const float* wT2 = (const float*)d_in[16];
  const float* bT2 = (const float*)d_in[17];
  float* out = (float*)d_out;

  const int NPAD = 8 * 128 * 1156;   // padded activation element count

  char* ws = (char*)d_ws;
  size_t off = 0;
  auto carve = [&](size_t bytes) { char* p = ws + off; off += (bytes + 255) & ~255ull; return p; };
  _Float16* feat_pad = (_Float16*)carve((size_t)NPAD * 2);   // [8*128][34][34]
  _Float16* hR_pad   = (_Float16*)carve((size_t)NPAD * 2);
  _Float16* hT_pad   = (_Float16*)carve((size_t)NPAD * 2);
  _Float16* wR1_16 = (_Float16*)carve(147456 * 2);           // [128][1152]
  _Float16* wT1_16 = (_Float16*)carve(147456 * 2);
  _Float16* wR2_16 = (_Float16*)carve(294912 * 2);           // [256][1152]
  _Float16* wT2_16 = (_Float16*)carve(55296 * 2);            // [48][1152]
  _Float16* r16    = (_Float16*)carve(2097152 * 2);          // [8][256][32][32]
  _Float16* t16    = (_Float16*)carve(393216 * 2);           // [8][48][32][32]
  int*      offt   = (int*)carve(1152 * 4);

  make_offt<<<5, 256, 0, stream>>>(offt);
  cvt_pad_f16<<<(NPAD + 255) / 256, 256, 0, stream>>>(features, feat_pad, NPAD);
  fill_zero_f16<<<(NPAD + 255) / 256, 256, 0, stream>>>(hR_pad, NPAD);
  fill_zero_f16<<<(NPAD + 255) / 256, 256, 0, stream>>>(hT_pad, NPAD);

  auto cvt = [&](const float* s, _Float16* d, int n) {
    cvt_f32_f16<<<(n + 255) / 256, 256, 0, stream>>>(s, d, n);
  };
  cvt(wR1, wR1_16, 147456);
  cvt(wT1, wT1_16, 147456);
  cvt(wR2, wR2_16, 294912);
  cvt(wT2, wT2_16, 55296);

  conv1_bn_silu<<<128, 256, 0, stream>>>(feat_pad, wR1_16, offt, bR1, gR, beR, mR, vR, hR_pad);
  conv1_bn_silu<<<128, 256, 0, stream>>>(feat_pad, wT1_16, offt, bT1, gT, beT, mT, vT, hT_pad);

  conv2_bias<<<dim3(128, 2), 256, 0, stream>>>(hR_pad, wR2_16, offt, bR2, r16, 256);
  conv2_bias<<<dim3(128, 1), 256, 0, stream>>>(hT_pad, wT2_16, offt, bT2, t16, 48);

  const size_t smem = 67584 + 16384 + 36864 + 4096 + 2048;   // 126976 B
  attention_cell<<<8 * 31 * 31, 256, smem, stream>>>(f, r16, t16, out);
}
